// TransformerBlock_46213848105804
// MI455X (gfx1250) — compile-verified, matrix operand order checked by reading, not run on hardware
//
#include <hip/hip_runtime.h>
#include <hip/hip_bf16.h>
#include <math.h>

typedef unsigned short u16;
typedef __attribute__((ext_vector_type(16))) __bf16 v16bf;
typedef __attribute__((ext_vector_type(8)))  float  v8f;

#define Bb 2
#define Lq 2048
#define Em 1024
#define Hh 16
#define Dh 64
#define Ff 4096
#define Mrows (Bb*Lq)

// padded LDS row stride (u16 units) for 32-wide k-tiles: 40 -> conflict-free b128 reads
#define LPAD 40

__device__ __forceinline__ u16 f32_to_bf16(float f){
  union { float f; unsigned u; } c; c.f = f;
  unsigned u = c.u;
  return (u16)((u + 0x7FFFu + ((u >> 16) & 1u)) >> 16);
}

union ABFrag { v16bf v; uint4 q[2]; };

__device__ __forceinline__ void load_frag_g(ABFrag& f, const u16* __restrict__ base){
  f.q[0] = *(const uint4*)base;
  f.q[1] = *(const uint4*)(base + 16);
}

// async copy of 16B from global to LDS (per lane), tracked by ASYNCcnt
__device__ __forceinline__ void async_cp16(unsigned lds_off, const u16* g){
  asm volatile("global_load_async_to_lds_b128 %0, %1, off"
               :: "v"(lds_off), "v"((unsigned long long)(uintptr_t)g)
               : "memory");
}

// ---------------------------------------------------------------------------
// Weight convert + transpose (LDS-tiled, coalesced both ways):
//   w [K][N] f32  ->  wt [N][K] bf16
// grid: (N/32, K/32), block: 256 = 32x8
// ---------------------------------------------------------------------------
__global__ __launch_bounds__(256) void tb_cvt_transpose(
    const float* __restrict__ w, u16* __restrict__ wt, int K, int N)
{
  __shared__ float tile[32][33];
  const int tx = threadIdx.x & 31;
  const int ty = threadIdx.x >> 5;       // 0..7
  const int k0 = blockIdx.y * 32;
  const int n0 = blockIdx.x * 32;
  #pragma unroll
  for (int r = 0; r < 32; r += 8)
    tile[r + ty][tx] = w[(size_t)(k0 + r + ty) * N + n0 + tx];
  __syncthreads();
  #pragma unroll
  for (int r = 0; r < 32; r += 8)
    wt[(size_t)(n0 + r + ty) * K + k0 + tx] = f32_to_bf16(tile[tx][r + ty]);
}

// ---------------------------------------------------------------------------
// LayerNorm (fp32 -> bf16), one 256-thread block per row of 1024 elements
// ---------------------------------------------------------------------------
__global__ __launch_bounds__(256) void tb_layernorm_bf16(
    const float* __restrict__ x, const float* __restrict__ w,
    const float* __restrict__ b, u16* __restrict__ out)
{
  const int row = blockIdx.x;
  const int t   = threadIdx.x;
  const float4 v = ((const float4*)(x + (size_t)row * Em))[t];

  __shared__ float red[8];
  float s = v.x + v.y + v.z + v.w;
  #pragma unroll
  for (int o = 16; o > 0; o >>= 1) s += __shfl_xor(s, o, 32);
  if ((t & 31) == 0) red[t >> 5] = s;
  __syncthreads();
  float tot = 0.f;
  #pragma unroll
  for (int i = 0; i < 8; i++) tot += red[i];
  const float mu = tot * (1.0f / (float)Em);

  float4 d; d.x = v.x - mu; d.y = v.y - mu; d.z = v.z - mu; d.w = v.w - mu;
  float s2 = d.x*d.x + d.y*d.y + d.z*d.z + d.w*d.w;
  __syncthreads();
  #pragma unroll
  for (int o = 16; o > 0; o >>= 1) s2 += __shfl_xor(s2, o, 32);
  if ((t & 31) == 0) red[t >> 5] = s2;
  __syncthreads();
  float tot2 = 0.f;
  #pragma unroll
  for (int i = 0; i < 8; i++) tot2 += red[i];
  const float rstd = rsqrtf(tot2 * (1.0f / (float)Em) + 1e-5f);

  const float4 wv = ((const float4*)w)[t];
  const float4 bv = ((const float4*)b)[t];
  union { u16 h[4]; uint2 u; } pk;
  pk.h[0] = f32_to_bf16(d.x * rstd * wv.x + bv.x);
  pk.h[1] = f32_to_bf16(d.y * rstd * wv.y + bv.y);
  pk.h[2] = f32_to_bf16(d.z * rstd * wv.z + bv.z);
  pk.h[3] = f32_to_bf16(d.w * rstd * wv.w + bv.w);
  *(uint2*)(out + (size_t)row * Em + t * 4) = pk.u;
}

// ---------------------------------------------------------------------------
// Tiled bf16 WMMA GEMM, triple-buffered async-LDS pipeline (2 stages in flight),
// with peeled steady-state loop (no per-iteration branch ladder / modulo).
//   C = A[M][K] * Bt[N][K]^T + bias, epilogue by mode.
// 8 waves (4Mx2N); block tile 128x128; wave tile 32x64 (2x4 C-frags); K-step 32.
// mode 0: bf16 scatter to [B,H,L,DH]    (q, k)
// mode 1: bf16 scatter to [B,H,DH,L]    (v transposed)
// mode 2: f32  out = res + acc + bias   (out-proj hidden, FFN down)
// mode 3: bf16 out = gelu(acc + bias)   (FFN up)
// ---------------------------------------------------------------------------
__global__ __launch_bounds__(256) void tb_gemm_bf16(
    const u16* __restrict__ A, const u16* __restrict__ Bt,
    const float* __restrict__ bias, const float* __restrict__ res,
    void* __restrict__ outp, int M, int N, int K, int mode)
{
  __shared__ __align__(16) u16 lA[3][128 * LPAD];
  __shared__ __align__(16) u16 lB[3][128 * LPAD];

  const int tid  = threadIdx.x;
  const int wave = tid >> 5;
  const int lane = tid & 31;
  const int wm   = wave >> 1;                 // 0..3
  const int wn   = wave & 1;                  // 0..1
  const int blockM = blockIdx.y * 128;
  const int blockN = blockIdx.x * 128;
  const int ml = lane & 15;
  const int kb = (lane >> 4) * 8;

  // copy work split: A = 512 chunks of 16B, B = 512 chunks of 16B
  const int r0 = tid >> 2;                    // row 0..63
  const int r1 = r0 + 64;                     // row 64..127
  const int cc = (tid & 3) * 8;               // u16 col offset in k-tile

  auto stage = [&](int k0, int buf){
    async_cp16((unsigned)(uintptr_t)&lA[buf][r0 * LPAD + cc],
               A  + (size_t)(blockM + r0) * K + k0 + cc);
    async_cp16((unsigned)(uintptr_t)&lA[buf][r1 * LPAD + cc],
               A  + (size_t)(blockM + r1) * K + k0 + cc);
    async_cp16((unsigned)(uintptr_t)&lB[buf][r0 * LPAD + cc],
               Bt + (size_t)(blockN + r0) * K + k0 + cc);
    async_cp16((unsigned)(uintptr_t)&lB[buf][r1 * LPAD + cc],
               Bt + (size_t)(blockN + r1) * K + k0 + cc);
  };

  v8f acc[2][4] = {};

  auto compute = [&](int buf){
    ABFrag a[2], b[4];
    #pragma unroll
    for (int i = 0; i < 2; i++) {
      const u16* ap = &lA[buf][(wm*32 + i*16 + ml) * LPAD + kb];
      a[i].q[0] = *(const uint4*)ap;  a[i].q[1] = *(const uint4*)(ap + 16);
    }
    #pragma unroll
    for (int j = 0; j < 4; j++) {
      const u16* bp = &lB[buf][(wn*64 + j*16 + ml) * LPAD + kb];
      b[j].q[0] = *(const uint4*)bp;  b[j].q[1] = *(const uint4*)(bp + 16);
    }
    #pragma unroll
    for (int i = 0; i < 2; i++)
      #pragma unroll
      for (int j = 0; j < 4; j++)
        acc[i][j] = __builtin_amdgcn_wmma_f32_16x16x32_bf16(
            false, a[i].v, false, b[j].v, (short)0, acc[i][j], false, false);
  };

  const int nk = K >> 5;                      // K multiple of 32, nk >= 2 here
  stage(0, 0);
  stage(32, 1);

  int cur = 0, sbuf = 2;
  // steady state: keep 2 stages in flight
  for (int t = 0; t + 2 < nk; ++t) {
    stage((t + 2) << 5, sbuf);
    asm volatile("s_wait_asynccnt 0x8" ::: "memory");   // stage t complete
    __syncthreads();
    compute(cur);
    __syncthreads();                                    // buf[cur] free to restage
    cur  = (cur  == 2) ? 0 : cur  + 1;
    sbuf = (sbuf == 2) ? 0 : sbuf + 1;
  }
  // drain: second-to-last stage
  asm volatile("s_wait_asynccnt 0x4" ::: "memory");
  __syncthreads();
  compute(cur);
  __syncthreads();
  cur = (cur == 2) ? 0 : cur + 1;
  // drain: last stage
  asm volatile("s_wait_asynccnt 0x0" ::: "memory");
  __syncthreads();
  compute(cur);

  const int rofs = (lane >> 4) * 8;
  const int nl   = lane & 15;
  #pragma unroll
  for (int i = 0; i < 2; i++)
  #pragma unroll
  for (int j = 0; j < 4; j++)
  #pragma unroll
  for (int r = 0; r < 8; r++) {
    const int row = blockM + wm*32 + i*16 + rofs + r;
    const int col = blockN + wn*64 + j*16 + nl;
    float v = acc[i][j][r] + bias[col];
    if (mode == 0) {
      u16* o = (u16*)outp;
      int bb = row >> 11, l = row & (Lq - 1);
      int h  = col >> 6,  d = col & (Dh - 1);
      o[((((size_t)bb*Hh + h)*Lq + l) << 6) + d] = f32_to_bf16(v);
    } else if (mode == 1) {
      u16* o = (u16*)outp;
      int bb = row >> 11, l = row & (Lq - 1);
      int h  = col >> 6,  d = col & (Dh - 1);
      o[(((size_t)bb*Hh + h)*Dh + d)*Lq + l] = f32_to_bf16(v);
    } else if (mode == 2) {
      float* o = (float*)outp;
      size_t idx = (size_t)row * N + col;
      o[idx] = res[idx] + v;
    } else {
      u16* o = (u16*)outp;
      float g = 0.5f * v * (1.0f + erff(v * 0.70710678118654752f));
      o[(size_t)row * N + col] = f32_to_bf16(g);
    }
  }
}

// ---------------------------------------------------------------------------
// Flash-style attention: one wave per (b*h, 16-query block).
// q,k: [B,H,L,DH] bf16.  vT: [B,H,DH,L] bf16.  ctx out: [B,L,E] bf16.
// S = Q K^T * 8.0, online softmax over key chunks of 32, O += P V.
// ---------------------------------------------------------------------------
__global__ __launch_bounds__(32) void tb_attention(
    const u16* __restrict__ q, const u16* __restrict__ k,
    const u16* __restrict__ vT, u16* __restrict__ ctx)
{
  const int bh   = blockIdx.x >> 7;          // 128 query blocks per (b,h)
  const int qb   = blockIdx.x & 127;
  const int lane = threadIdx.x;
  const int ml   = lane & 15;
  const int kb   = (lane >> 4) * 8;
  const int rofs = (lane >> 4) * 8;
  const int nl   = lane & 15;

  const u16* qbh = q  + (size_t)bh * Lq * Dh;
  const u16* kbh = k  + (size_t)bh * Lq * Dh;
  const u16* vbh = vT + (size_t)bh * Dh * Lq;

  ABFrag qa[2];
  #pragma unroll
  for (int i = 0; i < 2; i++)
    load_frag_g(qa[i], qbh + (size_t)(qb*16 + ml) * Dh + i*32 + kb);

  v8f o[4] = {};
  float mrow = -1e30f, lrow = 0.0f;          // live in lanes 0..15

  __shared__ __align__(16) float s_s[16 * 32];
  __shared__ __align__(16) u16   s_p[16 * 32];
  __shared__ float s_alpha[16];
  __shared__ float s_linv[16];

  for (int j0 = 0; j0 < Lq; j0 += 32) {
    v8f c0 = {}, c1 = {};
    #pragma unroll
    for (int kk = 0; kk < 2; kk++) {
      ABFrag b0, b1;
      load_frag_g(b0, kbh + (size_t)(j0      + ml) * Dh + kk*32 + kb);
      load_frag_g(b1, kbh + (size_t)(j0 + 16 + ml) * Dh + kk*32 + kb);
      c0 = __builtin_amdgcn_wmma_f32_16x16x32_bf16(false, qa[kk].v, false, b0.v, (short)0, c0, false, false);
      c1 = __builtin_amdgcn_wmma_f32_16x16x32_bf16(false, qa[kk].v, false, b1.v, (short)0, c1, false, false);
    }
    #pragma unroll
    for (int r = 0; r < 8; r++) {
      s_s[(r + rofs) * 32 + nl]      = c0[r] * 8.0f;   // ATTN_SCALE = sqrt(64)
      s_s[(r + rofs) * 32 + nl + 16] = c1[r] * 8.0f;
    }
    asm volatile("s_wait_dscnt 0" ::: "memory");

    if (lane < 16) {
      const float4* sr = (const float4*)(s_s + lane * 32);
      float4 t[8];
      float mx = -1e30f;
      #pragma unroll
      for (int i = 0; i < 8; i++) {
        t[i] = sr[i];
        mx = fmaxf(mx, fmaxf(fmaxf(t[i].x, t[i].y), fmaxf(t[i].z, t[i].w)));
      }
      const float mnew = fmaxf(mrow, mx);
      const float al   = __expf(mrow - mnew);
      float sum = 0.0f;
      u16* pr = s_p + lane * 32;
      #pragma unroll
      for (int i = 0; i < 8; i++) {
        float p0 = __expf(t[i].x - mnew), p1 = __expf(t[i].y - mnew);
        float p2 = __expf(t[i].z - mnew), p3 = __expf(t[i].w - mnew);
        sum += p0 + p1 + p2 + p3;
        union { u16 h[4]; uint2 u; } pk;
        pk.h[0] = f32_to_bf16(p0); pk.h[1] = f32_to_bf16(p1);
        pk.h[2] = f32_to_bf16(p2); pk.h[3] = f32_to_bf16(p3);
        *(uint2*)(pr + i * 4) = pk.u;
      }
      lrow = lrow * al + sum;
      mrow = mnew;
      s_alpha[lane] = al;
    }
    asm volatile("s_wait_dscnt 0" ::: "memory");

    float alv[8];
    #pragma unroll
    for (int r = 0; r < 8; r++) alv[r] = s_alpha[r + rofs];
    #pragma unroll
    for (int f = 0; f < 4; f++)
      #pragma unroll
      for (int r = 0; r < 8; r++) o[f][r] *= alv[r];

    ABFrag pa;
    pa.q[0] = *(const uint4*)(s_p + ml * 32 + kb);
    pa.q[1] = *(const uint4*)(s_p + ml * 32 + kb + 16);
    #pragma unroll
    for (int f = 0; f < 4; f++) {
      ABFrag vb;
      load_frag_g(vb, vbh + (size_t)(f*16 + ml) * Lq + j0 + kb);
      o[f] = __builtin_amdgcn_wmma_f32_16x16x32_bf16(false, pa.v, false, vb.v, (short)0, o[f], false, false);
    }
  }

  if (lane < 16) s_linv[lane] = 1.0f / lrow;
  asm volatile("s_wait_dscnt 0" ::: "memory");
  float li[8];
  #pragma unroll
  for (int r = 0; r < 8; r++) li[r] = s_linv[r + rofs];

  const int bb = bh >> 4, h = bh & 15;
  #pragma unroll
  for (int f = 0; f < 4; f++)
    #pragma unroll
    for (int r = 0; r < 8; r++) {
      const float vv = o[f][r] * li[r];
      const size_t row = (size_t)bb * Lq + qb*16 + r + rofs;
      ctx[row * Em + h * Dh + f*16 + nl] = f32_to_bf16(vv);
    }
}

// ---------------------------------------------------------------------------
// Host-side launcher
// ---------------------------------------------------------------------------
extern "C" void kernel_launch(void* const* d_in, const int* in_sizes, int n_in,
                              void* d_out, int out_size, void* d_ws, size_t ws_size,
                              hipStream_t stream) {
  (void)in_sizes; (void)n_in; (void)out_size; (void)ws_size;
  const float* x    = (const float*)d_in[0];
  const float* ln1w = (const float*)d_in[1];
  const float* ln1b = (const float*)d_in[2];
  const float* wq   = (const float*)d_in[3];
  const float* bq   = (const float*)d_in[4];
  const float* wk   = (const float*)d_in[5];
  const float* bk   = (const float*)d_in[6];
  const float* wv   = (const float*)d_in[7];
  const float* bv   = (const float*)d_in[8];
  const float* wo   = (const float*)d_in[9];
  const float* bo   = (const float*)d_in[10];
  const float* ln2w = (const float*)d_in[11];
  const float* ln2b = (const float*)d_in[12];
  const float* wu   = (const float*)d_in[13];
  const float* bu   = (const float*)d_in[14];
  const float* wd   = (const float*)d_in[15];
  const float* bd   = (const float*)d_in[16];
  float* out = (float*)d_out;

  char* ws = (char*)d_ws;
  size_t off = 0;
  auto alloc = [&](size_t bytes) -> void* {
    void* p = ws + off;
    off += (bytes + 255) & ~(size_t)255;
    return p;
  };
  u16*   xn     = (u16*)  alloc((size_t)Mrows * Em * 2);
  u16*   wqT    = (u16*)  alloc((size_t)Em * Em * 2);
  u16*   wkT    = (u16*)  alloc((size_t)Em * Em * 2);
  u16*   wvT    = (u16*)  alloc((size_t)Em * Em * 2);
  u16*   woT    = (u16*)  alloc((size_t)Em * Em * 2);
  u16*   wuT    = (u16*)  alloc((size_t)Em * Ff * 2);
  u16*   wdT    = (u16*)  alloc((size_t)Ff * Em * 2);
  u16*   qbuf   = (u16*)  alloc((size_t)Mrows * Em * 2);
  u16*   kbuf   = (u16*)  alloc((size_t)Mrows * Em * 2);
  u16*   vTbuf  = (u16*)  alloc((size_t)Mrows * Em * 2);
  u16*   ctx    = (u16*)  alloc((size_t)Mrows * Em * 2);
  float* hidden = (float*)alloc((size_t)Mrows * Em * 4);
  u16*   hn     = (u16*)  alloc((size_t)Mrows * Em * 2);
  u16*   up     = (u16*)  alloc((size_t)Mrows * Ff * 2);

  // 1) weights -> bf16 transposed (tiled)
  tb_cvt_transpose<<<dim3(Em/32, Em/32), 256, 0, stream>>>(wq, wqT, Em, Em);
  tb_cvt_transpose<<<dim3(Em/32, Em/32), 256, 0, stream>>>(wk, wkT, Em, Em);
  tb_cvt_transpose<<<dim3(Em/32, Em/32), 256, 0, stream>>>(wv, wvT, Em, Em);
  tb_cvt_transpose<<<dim3(Em/32, Em/32), 256, 0, stream>>>(wo, woT, Em, Em);
  tb_cvt_transpose<<<dim3(Ff/32, Em/32), 256, 0, stream>>>(wu, wuT, Em, Ff);
  tb_cvt_transpose<<<dim3(Em/32, Ff/32), 256, 0, stream>>>(wd, wdT, Ff, Em);

  // 2) LN1
  tb_layernorm_bf16<<<Mrows, 256, 0, stream>>>(x, ln1w, ln1b, xn);

  // 3) QKV projections
  dim3 gE(Em/128, Mrows/128);
  tb_gemm_bf16<<<gE, 256, 0, stream>>>(xn, wqT, bq, nullptr, qbuf,  Mrows, Em, Em, 0);
  tb_gemm_bf16<<<gE, 256, 0, stream>>>(xn, wkT, bk, nullptr, kbuf,  Mrows, Em, Em, 0);
  tb_gemm_bf16<<<gE, 256, 0, stream>>>(xn, wvT, bv, nullptr, vTbuf, Mrows, Em, Em, 1);

  // 4) attention
  tb_attention<<<Bb*Hh*(Lq/16), 32, 0, stream>>>(qbuf, kbuf, vTbuf, ctx);

  // 5) out-proj + residual -> hidden
  tb_gemm_bf16<<<gE, 256, 0, stream>>>(ctx, woT, bo, x, hidden, Mrows, Em, Em, 2);

  // 6) LN2
  tb_layernorm_bf16<<<Mrows, 256, 0, stream>>>(hidden, ln2w, ln2b, hn);

  // 7) FFN up (GELU)
  dim3 gF(Ff/128, Mrows/128);
  tb_gemm_bf16<<<gF, 256, 0, stream>>>(hn, wuT, bu, nullptr, up, Mrows, Ff, Em, 3);

  // 8) FFN down + residual -> out
  tb_gemm_bf16<<<gE, 256, 0, stream>>>(up, wdT, bd, hidden, out, Mrows, Em, Ff, 2);
}
